// RegularizedLoss_62955630624887
// MI455X (gfx1250) — compile-verified, matrix operand order checked by reading, not run on hardware
//
#include <hip/hip_runtime.h>

// ---------------------------------------------------------------------------
// RegularizedLoss: BCE-with-logits mean + 0.1 * einsum('bi,ij,bj->')/B
//   outputs: (16384,1000) f32   targets: (16384,1000) f32   co: (1000,1000) f32
// G = T^T O via v_wmma_f32_16x16x32_bf16 (fp32 accum); co tile prefetched to
// LDS with the Tensor Data Mover, fused elementwise contraction at the end.
// ---------------------------------------------------------------------------

typedef __attribute__((ext_vector_type(16))) __bf16        bf16x16;
typedef __attribute__((ext_vector_type(8)))  float         f32x8;
typedef __attribute__((ext_vector_type(4)))  unsigned int  u32x4;
typedef __attribute__((ext_vector_type(8)))  int           i32x8;
typedef __attribute__((ext_vector_type(4)))  int           i32x4;

#define B_DIM       16384
#define C_DIM       1000
#define LAMBDA_REG  0.1f

#define BCE_BLOCKS  2048
#define BCE_THREADS 256

#define KSPLIT       4
#define GEMM_THREADS 512        // 16 waves (wave32)
#define TILE         64         // 64x64 output tile per workgroup
#define KB           64         // K per stage = 2 WMMA k-steps
#define LDA          72         // padded LDS row stride (bf16), 144B, 16B-aligned
#define GEMM_BLOCKS  (16 * 16 * KSPLIT)

__device__ __forceinline__ unsigned short f2bf(float f) {
  // round-to-nearest-even fp32 -> bf16
  unsigned int u = __builtin_bit_cast(unsigned int, f);
  u += 0x7FFFu + ((u >> 16) & 1u);
  return (unsigned short)(u >> 16);
}

// ---------------------------------------------------------------------------
// Kernel 1: BCE-with-logits partial sums (streaming, memory bound)
// ---------------------------------------------------------------------------
__global__ void __launch_bounds__(BCE_THREADS)
bce_partial_kernel(const float* __restrict__ x, const float* __restrict__ t,
                   float* __restrict__ partials) {
  __shared__ float red[BCE_THREADS];
  const long long N = (long long)B_DIM * C_DIM;
  float acc = 0.f;
  for (long long n = (long long)blockIdx.x * blockDim.x + threadIdx.x; n < N;
       n += (long long)gridDim.x * blockDim.x) {
    float xv = x[n];
    float tv = t[n];
    acc += fmaxf(xv, 0.f) - xv * tv + log1pf(__expf(-fabsf(xv)));
  }
  red[threadIdx.x] = acc;
  __syncthreads();
  for (int s = BCE_THREADS / 2; s > 0; s >>= 1) {
    if ((int)threadIdx.x < s) red[threadIdx.x] += red[threadIdx.x + s];
    __syncthreads();
  }
  if (threadIdx.x == 0) partials[blockIdx.x] = red[0];
}

// ---------------------------------------------------------------------------
// Kernel 2: reg partials. Workgroup = 64x64 tile of G = T^T O, K split 4-way.
// 16 waves, each owns one 16x16 WMMA accumulator. TDM prefetches the
// co-occurrence tile into LDS overlapped with the whole K loop.
// ---------------------------------------------------------------------------
__global__ void __launch_bounds__(GEMM_THREADS)
reg_gemm_kernel(const float* __restrict__ targets,
                const float* __restrict__ outputs,
                const float* __restrict__ co,
                float* __restrict__ partials) {
  __shared__ unsigned short As[TILE][LDA];   // As[i_local][k_local]  (T^T tile)
  __shared__ unsigned short Bs[TILE][LDA];   // Bs[j_local][k_local]
  __shared__ float coTile[TILE * TILE];      // TDM destination (row-major 64x64)
  __shared__ float red[GEMM_THREADS];

  const int tid  = threadIdx.x;
  const int lane = tid & 31;
  const int wave = tid >> 5;      // 0..15
  const int wi   = wave >> 2;     // wave's row tile (0..3)
  const int wj   = wave & 3;      // wave's col tile (0..3)

  const int i0 = blockIdx.y * TILE;
  const int j0 = blockIdx.x * TILE;

  // ---- Tensor Data Mover: async co[i0:i0+64, j0:j0+64] -> LDS (wave 0) ----
  // D# per CDNA5 ISA. Tensor dims = remaining extent so
  // rows/cols past 1000 zero-fill; tile 64x64, dim0 stride 1000, 4B elements.
  if (wave == 0) {
    const unsigned long long ga =
        (unsigned long long)(uintptr_t)(co + (long long)i0 * C_DIM + j0);
    const unsigned int lds = (unsigned int)(uintptr_t)&coTile[0];
    const unsigned int td0 = (unsigned int)(C_DIM - j0);   // valid cols
    const unsigned int td1 = (unsigned int)(C_DIM - i0);   // valid rows

    u32x4 g0;
    g0[0] = 1u;                                            // count=1, valid
    g0[1] = lds;                                           // lds_addr
    g0[2] = (unsigned int)ga;                              // global_addr lo
    g0[3] = (unsigned int)((ga >> 32) & 0x01FFFFFFull) | (2u << 30);  // type=2

    i32x8 g1;
    g1[0] = (int)(2u << 16);                               // data_size=4B
    g1[1] = (int)((td0 & 0xFFFFu) << 16);                  // tensor_dim0 lo16
    g1[2] = (int)((td0 >> 16) | ((td1 & 0xFFFFu) << 16));  // td0 hi / td1 lo
    g1[3] = (int)((td1 >> 16) | (64u << 16));              // td1 hi / tile_dim0
    g1[4] = (int)64u;                                      // tile_dim1=64, tile_dim2=0
    g1[5] = (int)C_DIM;                                    // tensor_dim0_stride lo
    g1[6] = 0;                                             // stride hi / dim1_stride
    g1[7] = 0;

    const i32x4 z4 = {0, 0, 0, 0};
    const i32x8 z8 = {0, 0, 0, 0, 0, 0, 0, 0};
    // amdgpu-toolchain (clang-23) 6-arg form:
    // (u32x4 g0, i32x8 g1, i32x4, i32x4, i32x8, i32 cpol)
    __builtin_amdgcn_tensor_load_to_lds(g0, g1, z4, z4, z8, 0);
  }

  const int kStart = blockIdx.z * (B_DIM / KSPLIT);
  const int kIters = (B_DIM / KSPLIT) / KB;   // 64

  // staging: thread -> (k rows sk and sk+32, 4 consecutive columns sc..sc+3)
  const int sk = tid >> 4;          // 0..31
  const int sc = (tid & 15) * 4;    // 0,4,...,60
  // Edge tiles: clamp the column pointer into range so every float4 load is
  // in-bounds and UNCONDITIONAL (no exec-mask juggling in the hot loop).
  // Duplicated data lands only in G rows/cols >= C_DIM, which the guarded
  // tail contraction masks out.
  const int ci = min(i0 + sc, C_DIM - 4);
  const int cj = min(j0 + sc, C_DIM - 4);

  const float* tA = targets + (long long)(kStart + sk) * C_DIM + ci;
  const float* tB = outputs + (long long)(kStart + sk) * C_DIM + cj;
  const long long halfStep = 32LL * C_DIM;   // 32 k-rows
  const long long iterStep = (long long)KB * C_DIM;

  f32x8 acc = {};

  for (int it = 0; it < kIters; ++it) {
    __syncthreads();  // previous stage fully consumed

    if (it + 1 < kIters) {  // uniform branch; stay inside the arrays
      __builtin_prefetch(tA + iterStep, 0, 3);
      __builtin_prefetch(tB + iterStep, 0, 3);
    }

    #pragma unroll
    for (int h = 0; h < 2; ++h) {
      const int r = sk + 32 * h;
      const float4 va = *reinterpret_cast<const float4*>(tA + h * halfStep);
      const float4 vb = *reinterpret_cast<const float4*>(tB + h * halfStep);
      As[sc + 0][r] = f2bf(va.x);
      As[sc + 1][r] = f2bf(va.y);
      As[sc + 2][r] = f2bf(va.z);
      As[sc + 3][r] = f2bf(va.w);
      Bs[sc + 0][r] = f2bf(vb.x);
      Bs[sc + 1][r] = f2bf(vb.y);
      Bs[sc + 2][r] = f2bf(vb.z);
      Bs[sc + 3][r] = f2bf(vb.w);
    }
    __syncthreads();

    // CDNA5 16-bit A/B fragment layout:
    // lanes 0-15: K = {0..7, 16..23}; lanes 16-31: K = {8..15, 24..31}
    const int mrow = wi * 16 + (lane & 15);
    const int nrow = wj * 16 + (lane & 15);
    const int klo  = (lane & 16) ? 8 : 0;

    #pragma unroll
    for (int ks = 0; ks < 2; ++ks) {
      const __bf16* pa = reinterpret_cast<const __bf16*>(&As[mrow][ks * 32 + klo]);
      const __bf16* pb = reinterpret_cast<const __bf16*>(&Bs[nrow][ks * 32 + klo]);
      bf16x16 a, b;
      #pragma unroll
      for (int q = 0; q < 8; ++q) {
        a[q]     = pa[q];
        a[8 + q] = pa[16 + q];
        b[q]     = pb[q];
        b[8 + q] = pb[16 + q];
      }
      acc = __builtin_amdgcn_wmma_f32_16x16x32_bf16(
          /*neg_a=*/false, a, /*neg_b=*/false, b,
          /*c_mod=*/(short)0, acc, /*reuse_a=*/false, /*reuse_b=*/false);
    }

    tA += iterStep;
    tB += iterStep;
  }

  // ---- TDM completion, then fused contraction with co tile in LDS ----
  if (wave == 0) __builtin_amdgcn_s_wait_tensorcnt(0);
  __syncthreads();

  // C/D layout: VGPR r -> M = r + 8*(lane/16), N = lane%16.
  // Guard masks the clamped/duplicated G entries at i>=C_DIM or j>=C_DIM.
  float p = 0.f;
  #pragma unroll
  for (int r = 0; r < 8; ++r) {
    const int m = wi * 16 + r + 8 * (lane >> 4);
    const int n = wj * 16 + (lane & 15);
    if (i0 + m < C_DIM && j0 + n < C_DIM) p += acc[r] * coTile[m * TILE + n];
  }

  red[tid] = p;
  __syncthreads();
  for (int s = GEMM_THREADS / 2; s > 0; s >>= 1) {
    if (tid < s) red[tid] += red[tid + s];
    __syncthreads();
  }
  if (tid == 0) {
    const int bid = (blockIdx.z * gridDim.y + blockIdx.y) * gridDim.x + blockIdx.x;
    partials[bid] = red[0];
  }
}

// ---------------------------------------------------------------------------
// Kernel 3: deterministic final reduction (fixed order, no float atomics)
// ---------------------------------------------------------------------------
__global__ void __launch_bounds__(256)
finalize_kernel(const float* __restrict__ bce_partials,
                const float* __restrict__ reg_partials,
                float* __restrict__ out) {
  __shared__ float red[256];
  __shared__ float bce_total;

  float sb = 0.f;
  for (int i = threadIdx.x; i < BCE_BLOCKS; i += 256) sb += bce_partials[i];
  red[threadIdx.x] = sb;
  __syncthreads();
  for (int s = 128; s > 0; s >>= 1) {
    if ((int)threadIdx.x < s) red[threadIdx.x] += red[threadIdx.x + s];
    __syncthreads();
  }
  if (threadIdx.x == 0) bce_total = red[0];
  __syncthreads();

  float sr = 0.f;
  for (int i = threadIdx.x; i < GEMM_BLOCKS; i += 256) sr += reg_partials[i];
  red[threadIdx.x] = sr;
  __syncthreads();
  for (int s = 128; s > 0; s >>= 1) {
    if ((int)threadIdx.x < s) red[threadIdx.x] += red[threadIdx.x + s];
    __syncthreads();
  }
  if (threadIdx.x == 0) {
    const float base = bce_total / ((float)B_DIM * (float)C_DIM);
    const float reg  = red[0] / (float)B_DIM;
    out[0] = base + LAMBDA_REG * reg;
  }
}

// ---------------------------------------------------------------------------
extern "C" void kernel_launch(void* const* d_in, const int* in_sizes, int n_in,
                              void* d_out, int out_size, void* d_ws, size_t ws_size,
                              hipStream_t stream) {
  const float* outputs = (const float*)d_in[0];  // (B, C) logits
  const float* targets = (const float*)d_in[1];  // (B, C) soft labels
  const float* co      = (const float*)d_in[2];  // (C, C)

  float* bce_partials = (float*)d_ws;                 // [BCE_BLOCKS]
  float* reg_partials = bce_partials + BCE_BLOCKS;    // [GEMM_BLOCKS]

  bce_partial_kernel<<<BCE_BLOCKS, BCE_THREADS, 0, stream>>>(outputs, targets,
                                                             bce_partials);

  dim3 grid(16, 16, KSPLIT);  // 16*64 = 1024 >= 1000 in i and j
  reg_gemm_kernel<<<grid, GEMM_THREADS, 0, stream>>>(targets, outputs, co,
                                                     reg_partials);

  finalize_kernel<<<1, 256, 0, stream>>>(bce_partials, reg_partials,
                                         (float*)d_out);
}
